// SpaceProjection_25718264169214
// MI455X (gfx1250) — matrix operand order repaired and checked by reading, not compiled, not run on hardware
//
#include <hip/hip_runtime.h>

// ---------------- problem constants (match reference) ----------------
#define Bc   4
#define Cc   512
#define Nc   4096
#define C8c  64

typedef __attribute__((ext_vector_type(16))) __bf16 v16bf;
typedef __attribute__((ext_vector_type(8)))  __bf16 v8bf;
typedef __attribute__((ext_vector_type(4)))  __bf16 v4bf;
typedef __attribute__((ext_vector_type(8)))  float  v8f;
typedef __attribute__((ext_vector_type(4)))  float  v4f;

__device__ __forceinline__ v16bf cat16(v8bf lo, v8bf hi) {
    return __builtin_shufflevector(lo, hi, 0, 1, 2, 3, 4, 5, 6, 7,
                                           8, 9, 10, 11, 12, 13, 14, 15);
}

// A-frag (16x32 MxK, ISA 7.12.2): lane r=lane&15, hl=lane>>4 holds
// k in [8hl,8hl+8) then [16+8hl,+8) -> two contiguous 16B runs (row-major, ld%8==0).
__device__ __forceinline__ v16bf ld_afrag(const __bf16* p, int ld, int lane) {
    const int hl = (lane >> 4) & 1, r = lane & 15;
    const __bf16* q = p + r * ld + 8 * hl;
    return cat16(*(const v8bf*)q, *(const v8bf*)(q + 16));
}

// B-frag (32x16 KxN) from k-contiguous ("transposed") storage T[n][k]:
// lane n=lane&15, hl holds k in [8hl,8hl+8) then [16+8hl,+8) -> two contiguous 16B runs.
__device__ __forceinline__ v16bf ld_bfragT(const __bf16* p, int ld, int lane) {
    const int hl = (lane >> 4) & 1, n = lane & 15;
    const __bf16* q = p + n * ld + 8 * hl;
    return cat16(*(const v8bf*)q, *(const v8bf*)(q + 16));
}

// ---- CDNA5 async global->LDS copy (ASYNCcnt-tracked, ISA 10.x / 15.18.3) ----
__device__ __forceinline__ void async_ld16(void* lds_ptr, const void* gptr) {
    unsigned int       la = (unsigned int)(uintptr_t)lds_ptr;   // LDS offset = addr[31:0]
    unsigned long long ga = (unsigned long long)(uintptr_t)gptr;
    asm volatile("global_load_async_to_lds_b128 %0, %1, off"
                 :: "v"(la), "v"(ga) : "memory");
}
__device__ __forceinline__ void wait_async0() {
    asm volatile("s_wait_asynccnt 0x0" ::: "memory");
}

// ---------------- Kernel 1: fused QKV projection GEMM ----------------
// [Wq;Wk;Wv](640x512) @ x_b(512x4096) + bias -> bf16:
//   Qt[b][n][o]  (A-side row-major) | Kt[b][n][o] (k-contig B-side) | Vn[b][c][j]
__device__ __forceinline__ const float* wall_row(const float* Wq, const float* Wk,
                                                 const float* Wv, int r) {
    if (r < 64)  return Wq + (size_t)r * Cc;
    if (r < 128) return Wk + (size_t)(r - 64) * Cc;
    return Wv + (size_t)(r - 128) * Cc;
}

__global__ __launch_bounds__(256) void qkv_proj_kernel(
    const float* __restrict__ x,
    const float* __restrict__ Wq, const float* __restrict__ bq,
    const float* __restrict__ Wk, const float* __restrict__ bk,
    const float* __restrict__ Wv, const float* __restrict__ bv,
    __bf16* __restrict__ Qt, __bf16* __restrict__ Kt, __bf16* __restrict__ Vn)
{
    __shared__ __align__(16) __bf16 As[128 * 40];    // weights, row-major [r][k]
    __shared__ __align__(16) __bf16 BsT[128 * 40];   // x tile, transposed [n][k]

    const int tid  = threadIdx.x;
    const int lane = tid & 31;
    const int wave = tid >> 5;
    const int hl   = (lane >> 4) & 1;

    const int blk  = blockIdx.x;
    const int b    = blk / (5 * 32);
    const int rem  = blk % (5 * 32);
    const int row0 = (rem / 32) * 128;
    const int n0   = (rem % 32) * 128;

    const float* xb = x + (size_t)b * Cc * Nc;

    v8f acc[8] = {};

    for (int kt = 0; kt < Cc / 32; ++kt) {
        const int k0 = kt * 32;
        __syncthreads();
        // stage A (weights fp32 -> bf16), vectorized
        for (int t = tid * 4; t < 128 * 32; t += 1024) {
            int r = t >> 5, k = t & 31;
            v4f w = *(const v4f*)(wall_row(Wq, Wk, Wv, row0 + r) + k0 + k);
            v4bf pk;
            pk[0] = (__bf16)w[0]; pk[1] = (__bf16)w[1];
            pk[2] = (__bf16)w[2]; pk[3] = (__bf16)w[3];
            *(v4bf*)(As + r * 40 + k) = pk;
        }
        // stage B transposed (x fp32 -> bf16): coalesced v4f reads, LDS scatter
        for (int t = tid * 4; t < 32 * 128; t += 1024) {
            int k = t >> 7, n = t & 127;
            v4f xv = *(const v4f*)(xb + (size_t)(k0 + k) * Nc + n0 + n);
            BsT[(n + 0) * 40 + k] = (__bf16)xv[0];
            BsT[(n + 1) * 40 + k] = (__bf16)xv[1];
            BsT[(n + 2) * 40 + k] = (__bf16)xv[2];
            BsT[(n + 3) * 40 + k] = (__bf16)xv[3];
        }
        __syncthreads();

        // load all fragments first so dscnt waits pipeline across the 8 WMMAs
        v16bf afr = ld_afrag(As + wave * 16 * 40, 40, lane);
        v16bf bfr[8];
#pragma unroll
        for (int ct = 0; ct < 8; ++ct)
            bfr[ct] = ld_bfragT(BsT + ct * 16 * 40, 40, lane);
#pragma unroll
        for (int ct = 0; ct < 8; ++ct)
            acc[ct] = __builtin_amdgcn_wmma_f32_16x16x32_bf16(
                false, afr, false, bfr[ct], (short)0, acc[ct], false, false);
    }

    // epilogue: bias + bf16 + scatter (q/k rows contiguous per lane -> v8bf store)
    const int rbase = row0 + wave * 16 + 8 * hl;   // 8 consecutive output rows
#pragma unroll
    for (int ct = 0; ct < 8; ++ct) {
        const int gn = n0 + ct * 16 + (lane & 15);
        if (rbase < 64) {
            v8bf pk;
#pragma unroll
            for (int g = 0; g < 8; ++g) pk[g] = (__bf16)(acc[ct][g] + bq[rbase + g]);
            *(v8bf*)(Qt + ((size_t)b * Nc + gn) * C8c + rbase) = pk;
        } else if (rbase < 128) {
            v8bf pk;
#pragma unroll
            for (int g = 0; g < 8; ++g) pk[g] = (__bf16)(acc[ct][g] + bk[rbase - 64 + g]);
            *(v8bf*)(Kt + ((size_t)b * Nc + gn) * C8c + (rbase - 64)) = pk;
        } else {
#pragma unroll
            for (int g = 0; g < 8; ++g) {
                int c = rbase - 128 + g;
                Vn[((size_t)b * Cc + c) * Nc + gn] = (__bf16)(acc[ct][g] + bv[c]);
            }
        }
    }
}

// ---------------- Kernel 2: fused flash-style attention ----------------
// Block = 64 i-rows of one batch; O[64x512] fp32 in registers (8 waves x 16 v8f).
// V tiles double-buffered in LDS via async global->LDS copies (ASYNCcnt),
// prefetching chunk q+1 before computing chunk q.
#define SS_LD  68
#define PS_LD  72
#define VB_LD  40                       // 80B rows: 16B aligned, conflict-free frags
#define SM_AR  17408                    // Ss: 64*68*4 (Ps alias: 64*72*2 = 9216)
#define SM_MR  (SM_AR + 256)
#define SM_LR  (SM_MR + 256)
#define SM_MX  (SM_LR + 256)            // smax[4][64]
#define SM_SU  (SM_MX + 1024)           // ssum[4][64]
#define SM_VB0 (SM_SU + 1024)           // 20224, 16B aligned
#define SM_VB1 (SM_VB0 + 512 * VB_LD * 2)   // +40960 = 61184
#define SM_TOT (SM_VB1 + 512 * VB_LD * 2)   // 102144

__global__ __launch_bounds__(256) void attn_fused_kernel(
    const __bf16* __restrict__ Qt, const __bf16* __restrict__ Kt,
    const __bf16* __restrict__ Vn, const float* __restrict__ x,
    const float* __restrict__ gamma, float* __restrict__ out)
{
    __shared__ __align__(16) char smem[SM_TOT];
    float*  Ss   = (float*)(smem);
    __bf16* Ps   = (__bf16*)(smem);          // alias, phase-separated
    float*  arow = (float*)(smem + SM_AR);
    float*  mrow = (float*)(smem + SM_MR);
    float*  lrow = (float*)(smem + SM_LR);
    float*  smax = (float*)(smem + SM_MX);
    float*  ssum = (float*)(smem + SM_SU);
    __bf16* vb0  = (__bf16*)(smem + SM_VB0); // V chunk buffers: 512 c x 32 j
    __bf16* vb1  = (__bf16*)(smem + SM_VB1);

    const int tid  = threadIdx.x;
    const int lane = tid & 31;
    const int wave = tid >> 5;
    const int hl   = (lane >> 4) & 1;

    const int b  = blockIdx.x >> 6;
    const int i0 = (blockIdx.x & 63) * 64;

    const __bf16* Qtp = Qt + (size_t)b * Nc * C8c;
    const __bf16* Ktp = Kt + (size_t)b * Nc * C8c;
    const __bf16* Vnp = Vn + (size_t)b * Cc * Nc;

    // Q fragments (row subtile rq = wave>>1), kept in registers for all j-tiles.
    const int rq = wave >> 1;
    v16bf qf0 = ld_afrag(Qtp + (size_t)(i0 + rq * 16) * C8c,      C8c, lane);
    v16bf qf1 = ld_afrag(Qtp + (size_t)(i0 + rq * 16) * C8c + 32, C8c, lane);

    // O mapping: wave owns rows ri*16..+16, cols ch*256..+256.
    const int ri = wave & 3;
    const int ch = wave >> 2;
    v8f oacc[16] = {};

    float m_i = -1e30f, l_i = 0.0f;               // row state, threads tid<64
    const int srow = tid & 63, sseg = tid >> 6;   // softmax: 4 segments/row

    // issue async fill of one 512x32 V chunk (32KB; 8 x b128 per thread)
    auto issue_vfill = [&](int jbase, __bf16* vbuf) {
#pragma unroll
        for (int i = 0; i < 8; ++i) {
            int id = tid + i * 256;
            int c  = id >> 2;
            int jj = (id & 3) * 8;
            async_ld16(vbuf + c * VB_LD + jj, Vnp + (size_t)c * Nc + jbase + jj);
        }
    };

    issue_vfill(0, vb0);   // prefill chunk 0

    for (int jt = 0; jt < Nc / 64; ++jt) {
        const int j0 = jt * 64;

        __syncthreads();   // prior P/arow consumers done before Ss overwrite
        // ---- S = Q @ K_tile: 16 tiles over 8 waves x 2, B-frags from global ----
#pragma unroll
        for (int s = 0; s < 2; ++s) {
            const int cb = ((wave * 2 + s) & 3) * 16;
            v8f sv = {};
            v16bf kf0 = ld_bfragT(Ktp + (size_t)(j0 + cb) * C8c,      C8c, lane);
            sv = __builtin_amdgcn_wmma_f32_16x16x32_bf16(false, qf0, false, kf0,
                                                         (short)0, sv, false, false);
            v16bf kf1 = ld_bfragT(Ktp + (size_t)(j0 + cb) * C8c + 32, C8c, lane);
            sv = __builtin_amdgcn_wmma_f32_16x16x32_bf16(false, qf1, false, kf1,
                                                         (short)0, sv, false, false);
#pragma unroll
            for (int g = 0; g < 8; ++g)
                Ss[(rq * 16 + g + 8 * hl) * SS_LD + cb + (lane & 15)] = sv[g];
        }
        if (jt + 1 < Nc / 64)   // gfx1250 prefetch of next K tile
            __builtin_prefetch(Ktp + (size_t)(j0 + 64 + (tid & 63)) * C8c, 0, 1);
        __syncthreads();

        // ---- softmax phase A: each thread max-reduces its 16-wide segment ----
        v4f s0 = *(const v4f*)(Ss + srow * SS_LD + sseg * 16 + 0);
        v4f s1 = *(const v4f*)(Ss + srow * SS_LD + sseg * 16 + 4);
        v4f s2 = *(const v4f*)(Ss + srow * SS_LD + sseg * 16 + 8);
        v4f s3 = *(const v4f*)(Ss + srow * SS_LD + sseg * 16 + 12);
        float mx = -1e30f;
#pragma unroll
        for (int e = 0; e < 4; ++e)
            mx = fmaxf(mx, fmaxf(fmaxf(s0[e], s1[e]), fmaxf(s2[e], s3[e])));
        smax[sseg * 64 + srow] = mx;
        __syncthreads();

        // ---- phase B: row owners combine maxes, compute alpha ----
        if (tid < 64) {
            float mnew = fmaxf(fmaxf(smax[tid], smax[64 + tid]),
                               fmaxf(smax[128 + tid], smax[192 + tid]));
            mnew = fmaxf(m_i, mnew);
            arow[tid] = __expf(m_i - mnew);
            mrow[tid] = mnew;
            m_i = mnew;
        }
        __syncthreads();

        // ---- phase C: exp, write P (bf16, aliased), partial sums ----
        {
            const float mnew = mrow[srow];
            v8bf p0, p1;
            float sum = 0.0f;
#pragma unroll
            for (int e = 0; e < 4; ++e) { float p = __expf(s0[e] - mnew); sum += p; p0[e]     = (__bf16)p; }
#pragma unroll
            for (int e = 0; e < 4; ++e) { float p = __expf(s1[e] - mnew); sum += p; p0[4 + e] = (__bf16)p; }
#pragma unroll
            for (int e = 0; e < 4; ++e) { float p = __expf(s2[e] - mnew); sum += p; p1[e]     = (__bf16)p; }
#pragma unroll
            for (int e = 0; e < 4; ++e) { float p = __expf(s3[e] - mnew); sum += p; p1[4 + e] = (__bf16)p; }
            *(v8bf*)(Ps + srow * PS_LD + sseg * 16 + 0) = p0;
            *(v8bf*)(Ps + srow * PS_LD + sseg * 16 + 8) = p1;
            ssum[sseg * 64 + srow] = sum;
        }
        __syncthreads();

        // ---- phase D: row owners fold probability mass into l ----
        if (tid < 64) {
            l_i = arow[tid] * l_i +
                  (ssum[tid] + ssum[64 + tid] + ssum[128 + tid] + ssum[192 + tid]);
        }

        // ---- rescale O by alpha ----
        v4f a0 = *(const v4f*)(arow + ri * 16 + 8 * hl);
        v4f a1 = *(const v4f*)(arow + ri * 16 + 8 * hl + 4);
#pragma unroll
        for (int ct = 0; ct < 16; ++ct) {
#pragma unroll
            for (int g = 0; g < 8; ++g) {
                float al = (g < 4) ? a0[g & 3] : a1[g & 3];
                oacc[ct][g] *= al;
            }
        }

        // ---- O += P @ V^T: double-buffered async V chunks, LDS fragments ----
#pragma unroll
        for (int kc = 0; kc < 2; ++kc) {
            const int q = jt * 2 + kc;
            wait_async0();            // own chunk-q copies landed
            __syncthreads();          // everyone's copies landed; prev buf free
            if (q + 1 < (Nc / 64) * 2)
                issue_vfill((q + 1) * 32, (q & 1) ? vb0 : vb1);
            __bf16* vb = (q & 1) ? vb1 : vb0;
            v16bf pa = ld_afrag(Ps + ri * 16 * PS_LD + kc * 32, PS_LD, lane);
#pragma unroll
            for (int ct = 0; ct < 16; ++ct) {
                v16bf vfr = ld_bfragT(vb + (ch * 256 + ct * 16) * VB_LD, VB_LD, lane);
                oacc[ct] = __builtin_amdgcn_wmma_f32_16x16x32_bf16(
                    false, pa, false, vfr, (short)0, oacc[ct], false, false);
            }
        }
    }

    // ---- epilogue: out = gamma * O / l + x (vectorized, rows contiguous) ----
    if (tid < 64) lrow[tid] = l_i;
    __syncthreads();
    const float gm = gamma[0];
    const int rb = ri * 16 + 8 * hl;
    float inv[8];
#pragma unroll
    for (int g = 0; g < 8; ++g) inv[g] = gm / lrow[rb + g];
#pragma unroll
    for (int ct = 0; ct < 16; ++ct) {
        const int c = ch * 256 + ct * 16 + (lane & 15);
        const size_t base = ((size_t)b * Cc + c) * Nc + i0 + rb;
        v4f x0 = *(const v4f*)(x + base);
        v4f x1 = *(const v4f*)(x + base + 4);
        v4f o0, o1;
#pragma unroll
        for (int g = 0; g < 4; ++g) {
            o0[g] = oacc[ct][g]     * inv[g]     + x0[g];
            o1[g] = oacc[ct][g + 4] * inv[g + 4] + x1[g];
        }
        *(v4f*)(out + base)     = o0;
        *(v4f*)(out + base + 4) = o1;
    }
}

// ---------------- launcher ----------------
extern "C" void kernel_launch(void* const* d_in, const int* in_sizes, int n_in,
                              void* d_out, int out_size, void* d_ws, size_t ws_size,
                              hipStream_t stream) {
    const float* x     = (const float*)d_in[0];
    const float* Wq    = (const float*)d_in[1];
    const float* bq    = (const float*)d_in[2];
    const float* Wk    = (const float*)d_in[3];
    const float* bk    = (const float*)d_in[4];
    const float* Wv    = (const float*)d_in[5];
    const float* bv    = (const float*)d_in[6];
    const float* gamma = (const float*)d_in[7];
    float* out = (float*)d_out;

    // workspace (bf16): Qt[B][N][64] | Kt[B][N][64] | Vn[B][C][N]  (~20 MB)
    __bf16* Qt = (__bf16*)d_ws;
    __bf16* Kt = Qt + (size_t)Bc * Nc * C8c;
    __bf16* Vn = Kt + (size_t)Bc * Nc * C8c;

    qkv_proj_kernel<<<dim3(Bc * 5 * (Nc / 128)), dim3(256), 0, stream>>>(
        x, Wq, bq, Wk, bk, Wv, bv, Qt, Kt, Vn);

    attn_fused_kernel<<<dim3(Bc * (Nc / 64)), dim3(256), 0, stream>>>(
        Qt, Kt, Vn, x, gamma, out);
}
